// BlockwiseParallelTransformer_39075612459199
// MI455X (gfx1250) — matrix operand order repaired and checked by reading, not compiled, not run on hardware
//
#include <hip/hip_runtime.h>

typedef unsigned short u16;
typedef unsigned int   u32;
typedef __bf16 bf16_t;
typedef bf16_t v16bf __attribute__((ext_vector_type(16)));
typedef float  v8f   __attribute__((ext_vector_type(8)));
typedef int    v4i_vs __attribute__((__vector_size__(16)));   // matches builtin prototype

#define AS1 __attribute__((address_space(1)))
#define AS3 __attribute__((address_space(3)))

#if __has_builtin(__builtin_amdgcn_global_load_async_to_lds_b128)
#define HAVE_ASYNC_LDS 1
#else
#define HAVE_ASYNC_LDS 0
#endif

struct frag_raw { uint4 a, b; };
union  frag_u   { frag_raw r; v16bf v; };

__device__ __forceinline__ u16 f2bf(float f) {
    u32 u = __float_as_uint(f);
    u += 0x7FFFu + ((u >> 16) & 1u);   // round-to-nearest-even
    return (u16)(u >> 16);
}

__device__ __forceinline__ v8f wmma_bf16(v16bf a, v16bf b, v8f c) {
    return __builtin_amdgcn_wmma_f32_16x16x32_bf16(false, a, false, b, (short)0, c, false, false);
}

// 16B memory -> LDS mover: async DMA (ASYNCcnt) on gfx1250, sync copy fallback.
__device__ __forceinline__ void ld_lds_b128(u16* lds, const u16* g) {
#if HAVE_ASYNC_LDS
    __builtin_amdgcn_global_load_async_to_lds_b128(
        (AS1 v4i_vs*)(uintptr_t)g,
        (AS3 v4i_vs*)(u32)(uintptr_t)lds, 0, 0);
#else
    *(uint4*)lds = *(const uint4*)g;
#endif
}

__device__ __forceinline__ void async_wait_all() {
#if HAVE_ASYNC_LDS
#if __has_builtin(__builtin_amdgcn_s_wait_asynccnt)
    __builtin_amdgcn_s_wait_asynccnt(0);
#else
    asm volatile("s_wait_asynccnt 0" ::: "memory");
#endif
#endif
}

// A-operand fragment (16x32 bf16) from row-major [rows x K] LDS tile.
// lanes 0-15: row=l, K in {k0+0..7, k0+16..23}; lanes 16-31: K in {k0+8..15, k0+24..31}
__device__ __forceinline__ v16bf load_fragA(const u16* s, int stride, int row0, int k0, int lane) {
    int l = lane & 15, hi = lane >> 4;
    const u16* p = s + (row0 + l) * stride + k0 + hi * 8;
    frag_u u;
    u.r.a = *(const uint4*)p;
    u.r.b = *(const uint4*)(p + 16);
    return u.v;
}

// B-operand fragment (32x16 bf16) from row-major [N x K] LDS tile.
// lanes 0-15: col n=l, K in {k0+0..7, k0+16..23}; lanes 16-31: col n=l-16, K in {k0+8..15, k0+24..31}
__device__ __forceinline__ v16bf load_fragB(const u16* s, int stride, int row0, int k0, int lane) {
    int l = lane & 15, hi = lane >> 4;
    const u16* p = s + (row0 + l) * stride + k0 + hi * 8;
    frag_u u;
    u.r.a = *(const uint4*)p;
    u.r.b = *(const uint4*)(p + 16);
    return u.v;
}

// ---------------- elementwise kernels ----------------
__global__ void k_cvt(const float* __restrict__ in, u16* __restrict__ out, int n) {
    int i = blockIdx.x * blockDim.x + threadIdx.x;
    int st = gridDim.x * blockDim.x;
    for (; i < n; i += st) out[i] = f2bf(in[i]);
}

__global__ void k_addcvt(const float* __restrict__ a, const float* __restrict__ b,
                         float* __restrict__ h1f, u16* __restrict__ h1b, int n) {
    int i = blockIdx.x * blockDim.x + threadIdx.x;
    int st = gridDim.x * blockDim.x;
    for (; i < n; i += st) {
        float v = a[i] + b[i];
        h1f[i] = v;
        h1b[i] = f2bf(v);
    }
}

// ---------------- GEMM: C[M,N] = A[M,K] * B[N,K]^T (+bias, modes) ----------------
#define GLDS 72   // padded LDS row stride (bf16 elems): 144B rows -> 16B aligned, conflict-free

__global__ void __launch_bounds__(256)
k_gemm(const u16* __restrict__ A, const u16* __restrict__ B, const float* __restrict__ bias,
       u16* __restrict__ outB, float* __restrict__ outF, const float* __restrict__ extra,
       int M, int N, int K, int mode, float scale) {
    __shared__ u16 sA[128 * GLDS];
    __shared__ u16 sB[128 * GLDS];
    int nb = N >> 7;
    int tm = blockIdx.x / nb, tn = blockIdx.x % nb;
    int rowBase = tm << 7, colBase = tn << 7;
    int tid = threadIdx.x;
    int lane = tid & 31, w = tid >> 5;
    int wm = w >> 2, wn = w & 3;           // wave tile: rows [wm*64,+64), cols [wn*32,+32)

    v8f acc[4][2];
#pragma unroll
    for (int i = 0; i < 4; i++)
#pragma unroll
        for (int j = 0; j < 2; j++)
#pragma unroll
            for (int e = 0; e < 8; e++) acc[i][j][e] = 0.f;

    for (int kb = 0; kb < K; kb += 64) {
        __syncthreads();                     // all waves done reading previous tile
#pragma unroll
        for (int c = 0; c < 4; c++) {
            int chunk = c * 256 + tid;       // 1024 chunks of 8 bf16 per matrix
            int rr = chunk >> 3;
            int co = (chunk & 7) << 3;
            ld_lds_b128(sA + rr * GLDS + co, A + (size_t)(rowBase + rr) * K + kb + co);
            ld_lds_b128(sB + rr * GLDS + co, B + (size_t)(colBase + rr) * K + kb + co);
        }
        async_wait_all();
        __syncthreads();                     // publish tile to all waves
#pragma unroll
        for (int ks = 0; ks < 2; ks++) {
            v16bf af[4], bfr[2];
#pragma unroll
            for (int mt = 0; mt < 4; mt++) af[mt]  = load_fragA(sA, GLDS, (wm << 6) + (mt << 4), ks << 5, lane);
#pragma unroll
            for (int nt = 0; nt < 2; nt++) bfr[nt] = load_fragB(sB, GLDS, (wn << 5) + (nt << 4), ks << 5, lane);
#pragma unroll
            for (int mt = 0; mt < 4; mt++)
#pragma unroll
                for (int nt = 0; nt < 2; nt++)
                    acc[mt][nt] = wmma_bf16(af[mt], bfr[nt], acc[mt][nt]);
        }
    }
    int l16 = lane & 15, hi = lane >> 4;
#pragma unroll
    for (int mt = 0; mt < 4; mt++)
#pragma unroll
        for (int nt = 0; nt < 2; nt++) {
            int col = colBase + (wn << 5) + (nt << 4) + l16;
            float bv = bias[col];
#pragma unroll
            for (int r = 0; r < 8; r++) {
                int row = rowBase + (wm << 6) + (mt << 4) + (hi << 3) + r;
                float v = acc[mt][nt][r] + bv;
                size_t idx = (size_t)row * N + col;
                if (mode == 0)      outB[idx] = f2bf(v * scale);
                else if (mode == 1) outB[idx] = f2bf(v > 0.f ? v : 0.f);
                else                outF[idx] = v + extra[idx];
            }
        }
}

// ---------------- blockwise attention ----------------
// grid: 256 WGs = 16 h * 16 query-tiles; 256 threads = 8 waves.
// Wave w produces output row r = (qt*8+w)*16 + h, all 1024 columns.
__global__ void __launch_bounds__(256)
k_attn(const u16* __restrict__ Q, const u16* __restrict__ Km,
       const u16* __restrict__ Vm, float* __restrict__ attn) {
    __shared__ u16 sQ[128 * 72];       // 128 query-vectors x d=64
    __shared__ u16 sK[64 * 72];        // 64 keys x d=64 (half block)
    __shared__ u16 sVt[2][64 * 72];    // V^T: f=64 rows x 64 keys, double-buffered per half
    __shared__ u16 sE[128 * 72];       // exp'ed scores: 128 queries x 64 keys

    int h  = blockIdx.x >> 4;
    int qt = blockIdx.x & 15;
    int tid = threadIdx.x;
    int lane = tid & 31, w = tid >> 5;
    int l16 = lane & 15, hi = lane >> 4;

    // stage Q tile: Qh row g -> global row (qt*8 + g/16)*16 + h, cols (g%16)*64 + d
#pragma unroll
    for (int c = 0; c < 4; c++) {
        int chunk = c * 256 + tid;
        int g = chunk >> 3;
        int doff = (chunk & 7) << 3;
        int r = (qt * 8 + (g >> 4)) * 16 + h;
        int col = (g & 15) * 64 + doff;
        ld_lds_b128(sQ + g * 72 + doff, Q + (size_t)r * 1024 + col);
    }

    v8f num[4];
#pragma unroll
    for (int i = 0; i < 4; i++)
#pragma unroll
        for (int e = 0; e < 8; e++) num[i][e] = 0.f;
    float den8[8];
#pragma unroll
    for (int i = 0; i < 8; i++) den8[i] = 0.f;

    for (int Kb = 0; Kb < 16; Kb++) {
        v8f acc[2][4];
#pragma unroll
        for (int half = 0; half < 2; half++) {
            int s0 = Kb * 128 + half * 64;
            __syncthreads();                    // protect sK / sVt[half] reuse
#pragma unroll
            for (int c = 0; c < 2; c++) {       // K half: 64x64
                int chunk = c * 256 + tid;
                int s = chunk >> 3;
                int doff = (chunk & 7) << 3;
                ld_lds_b128(sK + s * 72 + doff,
                            Km + (size_t)(s0 + s) * 1024 + h * 64 + doff);
            }
#pragma unroll
            for (int c = 0; c < 4; c++) {       // V^T half: transpose on the way in
                int idx = c * 256 + tid;
                int s = idx >> 4;
                int f = (idx & 15) << 2;
                uint2 vv = *(const uint2*)(Vm + (size_t)(s0 + s) * 1024 + h * 64 + f);
                sVt[half][(f + 0) * 72 + s] = (u16)(vv.x & 0xFFFFu);
                sVt[half][(f + 1) * 72 + s] = (u16)(vv.x >> 16);
                sVt[half][(f + 2) * 72 + s] = (u16)(vv.y & 0xFFFFu);
                sVt[half][(f + 3) * 72 + s] = (u16)(vv.y >> 16);
            }
            async_wait_all();                   // sQ (first pass) + sK DMA complete
            __syncthreads();
#pragma unroll
            for (int nt = 0; nt < 4; nt++)
#pragma unroll
                for (int e = 0; e < 8; e++) acc[half][nt][e] = 0.f;
#pragma unroll
            for (int ks = 0; ks < 2; ks++) {    // S = Qtile(16x64) x Khalf^T(64x64)
                v16bf af = load_fragA(sQ, 72, w << 4, ks << 5, lane);
#pragma unroll
                for (int nt = 0; nt < 4; nt++) {
                    v16bf bfr = load_fragB(sK, 72, nt << 4, ks << 5, lane);
                    acc[half][nt] = wmma_bf16(af, bfr, acc[half][nt]);
                }
            }
        }
        // per-row LOCAL max over the full 128-key block (both halves)
        float mx[8];
#pragma unroll
        for (int r = 0; r < 8; r++) {
            float m = acc[0][0][r];
#pragma unroll
            for (int nt = 1; nt < 4; nt++) m = fmaxf(m, acc[0][nt][r]);
#pragma unroll
            for (int nt = 0; nt < 4; nt++) m = fmaxf(m, acc[1][nt][r]);
#pragma unroll
            for (int sh = 1; sh < 16; sh <<= 1) m = fmaxf(m, __shfl_xor(m, sh, 16));
            mx[r] = m;
        }
#pragma unroll
        for (int r = 0; r < 8; r++) {
            float s = 0.f;
#pragma unroll
            for (int half = 0; half < 2; half++)
#pragma unroll
                for (int nt = 0; nt < 4; nt++) {
                    float e = __expf(acc[half][nt][r] - mx[r]);
                    acc[half][nt][r] = e;
                    s += e;
                }
#pragma unroll
            for (int sh = 1; sh < 16; sh <<= 1) s += __shfl_xor(s, sh, 16);
            den8[r] += s;
        }
        // num += E(16x64) x Vhalf(64x64), per half
#pragma unroll
        for (int half = 0; half < 2; half++) {
            __syncthreads();                    // protect sE reuse
#pragma unroll
            for (int nt = 0; nt < 4; nt++)
#pragma unroll
                for (int r = 0; r < 8; r++)
                    sE[((w << 4) + (hi << 3) + r) * 72 + (nt << 4) + l16] = f2bf(acc[half][nt][r]);
            __syncthreads();
#pragma unroll
            for (int ks = 0; ks < 2; ks++) {
                v16bf ef = load_fragA(sE, 72, w << 4, ks << 5, lane);
#pragma unroll
                for (int ft = 0; ft < 4; ft++) {
                    v16bf vf = load_fragB(sVt[half], 72, ft << 4, ks << 5, lane);
                    num[ft] = wmma_bf16(ef, vf, num[ft]);
                }
            }
        }
    }
    // write: row r, col = q*64 + f with q = tile-row m, f = ft*16 + lane16
    int r = (qt * 8 + w) * 16 + h;
#pragma unroll
    for (int ft = 0; ft < 4; ft++)
#pragma unroll
        for (int rr = 0; rr < 8; rr++) {
            int m = (hi << 3) + rr;
            int col = m * 64 + (ft << 4) + l16;
            attn[(size_t)r * 1024 + col] = num[ft][rr] / den8[rr];
        }
}

// ---------------- host launcher ----------------
extern "C" void kernel_launch(void* const* d_in, const int* in_sizes, int n_in,
                              void* d_out, int out_size, void* d_ws, size_t ws_size,
                              hipStream_t stream) {
    (void)in_sizes; (void)n_in; (void)out_size; (void)ws_size;
    const float* x  = (const float*)d_in[0];
    const float* Wq = (const float*)d_in[1];
    const float* bq = (const float*)d_in[2];
    const float* Wk = (const float*)d_in[3];
    const float* bk = (const float*)d_in[4];
    const float* Wv = (const float*)d_in[5];
    const float* bv = (const float*)d_in[6];
    const float* W1 = (const float*)d_in[7];
    const float* b1 = (const float*)d_in[8];
    const float* W2 = (const float*)d_in[9];
    const float* b2 = (const float*)d_in[10];
    float* out = (float*)d_out;

    const int N_ = 2048, D_ = 1024, FF_ = 4096;
    char* ws = (char*)d_ws;
    size_t off = 0;
    auto alloc = [&](size_t bytes) -> void* {
        void* p = ws + off;
        off += (bytes + 255) & ~(size_t)255;
        return p;
    };

    u16*   xb    = (u16*)alloc((size_t)N_ * D_ * 2);
    u16*   Wqb   = (u16*)alloc((size_t)D_ * D_ * 2);
    u16*   Wkb   = (u16*)alloc((size_t)D_ * D_ * 2);
    u16*   Wvb   = (u16*)alloc((size_t)D_ * D_ * 2);
    u16*   W1b   = (u16*)alloc((size_t)FF_ * D_ * 2);
    u16*   W2b   = (u16*)alloc((size_t)D_ * FF_ * 2);
    u16*   qb    = (u16*)alloc((size_t)N_ * D_ * 2);
    u16*   kbuf  = (u16*)alloc((size_t)N_ * D_ * 2);
    u16*   vbuf  = (u16*)alloc((size_t)N_ * D_ * 2);
    float* attnF = (float*)alloc((size_t)N_ * D_ * 4);
    float* h1f   = (float*)alloc((size_t)N_ * D_ * 4);
    u16*   h1b   = (u16*)alloc((size_t)N_ * D_ * 2);
    u16*   hb    = (u16*)alloc((size_t)N_ * FF_ * 2);

    // bf16 conversions
    k_cvt<<<1024, 256, 0, stream>>>(x,  xb,  N_ * D_);
    k_cvt<<<1024, 256, 0, stream>>>(Wq, Wqb, D_ * D_);
    k_cvt<<<1024, 256, 0, stream>>>(Wk, Wkb, D_ * D_);
    k_cvt<<<1024, 256, 0, stream>>>(Wv, Wvb, D_ * D_);
    k_cvt<<<1024, 256, 0, stream>>>(W1, W1b, FF_ * D_);
    k_cvt<<<1024, 256, 0, stream>>>(W2, W2b, D_ * FF_);

    // QKV projections (q scaled by 1/sqrt(64))
    int gQKV = (N_ / 128) * (D_ / 128);
    k_gemm<<<gQKV, 256, 0, stream>>>(xb, Wqb, bq, qb,   nullptr, nullptr, N_, D_, D_, 0, 0.125f);
    k_gemm<<<gQKV, 256, 0, stream>>>(xb, Wkb, bk, kbuf, nullptr, nullptr, N_, D_, D_, 0, 1.0f);
    k_gemm<<<gQKV, 256, 0, stream>>>(xb, Wvb, bv, vbuf, nullptr, nullptr, N_, D_, D_, 0, 1.0f);

    // blockwise attention
    k_attn<<<256, 256, 0, stream>>>(qb, kbuf, vbuf, attnF);

    // h1 = attn + x (f32 kept for residual, bf16 for FFN input)
    k_addcvt<<<1024, 256, 0, stream>>>(attnF, x, h1f, h1b, N_ * D_);

    // FFN
    k_gemm<<<(N_ / 128) * (FF_ / 128), 256, 0, stream>>>(h1b, W1b, b1, hb, nullptr, nullptr,
                                                         N_, FF_, D_, 1, 1.0f);
    // out = relu(...)@W2^T + b2 + h1f   (== y + attn + x)
    k_gemm<<<(N_ / 128) * (D_ / 128), 256, 0, stream>>>(hb, W2b, b2, nullptr, out, h1f,
                                                        N_, D_, FF_, 2, 1.0f);
}
